// PolaLinearAttention_33182917328967
// MI455X (gfx1250) — matrix operand order repaired and checked by reading, not compiled, not run on hardware
//
#include <hip/hip_runtime.h>
#include <hip/hip_bf16.h>
#include <math.h>

// ---- Problem constants -----------------------------------------------------
#define BB 32
#define NN 512
#define CC 384
#define HH 6
#define HD 64
#define M_TOT (BB * NN)      // 16384
#define QKV_COLS (4 * CC)    // 1536

typedef __attribute__((ext_vector_type(16))) _Float16 v16h;
typedef __attribute__((ext_vector_type(8)))  _Float16 v8h;
typedef __attribute__((ext_vector_type(4)))  _Float16 v4h;
typedef __attribute__((ext_vector_type(2)))  __fp16   h2raw;   // cvt_pkrtz result type
typedef __attribute__((ext_vector_type(8)))  float    v8f;
typedef __attribute__((ext_vector_type(4)))  float    f32x4;
typedef __attribute__((ext_vector_type(4)))  int      i32x4;

// Tile sizes for the WMMA GEMMs
#define TM 128   // rows per block (8 waves x 16)
#define TN 64    // cols per block (4 wmma subtiles of 16)
#define TK 32    // k per step (one wmma K)
#define LDH 48   // LDS half-stride per row (96B, keeps 16B chunk alignment)

__device__ __forceinline__ float sigmoid_f(float x) { return 1.0f / (1.0f + expf(-x)); }
__device__ __forceinline__ float softplus_f(float x) { return log1pf(expf(x)); }

// f32x4 -> 4 packed halves (v_cvt_pk_rtz_f16_f32 x2)
__device__ __forceinline__ v4h cvt4_f16(f32x4 f) {
    union { v4h h; h2raw p[2]; } u;
    u.p[0] = __builtin_amdgcn_cvt_pkrtz(f.x, f.y);
    u.p[1] = __builtin_amdgcn_cvt_pkrtz(f.z, f.w);
    return u.h;
}

// ---- Async global->LDS copy (CDNA5 GLOBAL_LOAD_ASYNC_TO_LDS_B128) ----------
#if defined(__has_builtin)
#if __has_builtin(__builtin_amdgcn_global_load_async_to_lds_b128) && \
    __has_builtin(__builtin_amdgcn_s_wait_asynccnt)
#define USE_ASYNC_LDS 1
#endif
#endif
#ifndef USE_ASYNC_LDS
#define USE_ASYNC_LDS 0
#endif

// Copy 16 bytes (4 floats) global -> LDS. Pointers must be 16B aligned.
__device__ __forceinline__ void async_copy_b128(const float* g, float* l) {
#if USE_ASYNC_LDS
    // builtin signature: (v4i addrspace(1)*, v4i addrspace(3)*, imm offset, imm cpol)
    // flat LDS pointer low 32 bits == LDS offset (ISA 10.2 aperture mapping)
    __builtin_amdgcn_global_load_async_to_lds_b128(
        (__attribute__((address_space(1))) i32x4*)(uintptr_t)g,
        (__attribute__((address_space(3))) i32x4*)(uintptr_t)l, 0, 0);
#else
    *(f32x4*)l = *(const f32x4*)g;
#endif
}
__device__ __forceinline__ void async_copy_wait() {
#if USE_ASYNC_LDS
    __builtin_amdgcn_s_wait_asynccnt(0);
#endif
}

// Load A fragment (16x32 f16) per ISA table: lanes 0-15 row M=lane, K in
// {0..7,16..23}; lanes 16-31 row M=lane-16, K in {8..15,24..31}.
__device__ __forceinline__ v16h load_a_frag(const _Float16* lds, int rowBase, int lane) {
    int r = rowBase + (lane & 15);
    int kb = (lane >> 4) << 3;   // 0 or 8
    union { v16h v; v8h h[2]; } u;
    u.h[0] = *(const v8h*)&lds[r * LDH + kb];
    u.h[1] = *(const v8h*)&lds[r * LDH + kb + 16];
    return u.v;
}

// Load B fragment (32x16 f16, stored N-major as [n][k]): lanes 0-15 col
// N=lane with K {0..7,16..23}, lanes 16-31 col N=lane-16 with K {8..15,24..31}.
__device__ __forceinline__ v16h load_b_frag(const _Float16* lds, int nBase, int lane) {
    int n = nBase + (lane & 15);
    int kb = (lane >> 4) << 3;   // 0 or 8
    union { v16h v; v8h h[2]; } u;
    u.h[0] = *(const v8h*)&lds[n * LDH + kb];
    u.h[1] = *(const v8h*)&lds[n * LDH + kb + 16];
    return u.v;
}

// ---------------------------------------------------------------------------
// K1: fused QKV GEMM.  out cols [0,384)=q from y, [384,768)=k, [768,1152)=v,
// [1152,1536)=g from x.  Epilogue applies bias (+pe for k), 1/softplus scale,
// powered-relu feature maps, and scatters into head-major feature buffers.
// ---------------------------------------------------------------------------
__global__ __launch_bounds__(256)
void qkv_gemm_kernel(const float* __restrict__ x, const float* __restrict__ y,
                     const float* __restrict__ W, const float* __restrict__ bias,
                     const float* __restrict__ power_p, const float* __restrict__ scale_p,
                     const float* __restrict__ pe,
                     float* __restrict__ q_feat, float* __restrict__ k_feat,
                     float* __restrict__ v_buf, float* __restrict__ g_buf) {
    __shared__ _Float16 As[TM * LDH];
    __shared__ _Float16 Bs[TN * LDH];

    const int t = threadIdx.x;
    const int lane = t & 31;
    const int wave = t >> 5;
    const int rowBase = blockIdx.y * TM;
    const int colBase = blockIdx.x * TN;
    const int cg = colBase / CC;                 // 0=q 1=k 2=v 3=g
    const float* A = (cg == 0) ? y : x;

    v8f acc[4] = {};

    for (int k0 = 0; k0 < CC; k0 += TK) {
        // stage A tile 128x32: float4 loads, packed cvt, b64 LDS stores
        #pragma unroll
        for (int i = 0; i < 4; ++i) {
            int e = t + i * 256;          // 1024 float4 slots
            int r = e >> 3, kq = (e & 7) * 4;
            f32x4 f = *(const f32x4*)&A[(size_t)(rowBase + r) * CC + k0 + kq];
            *(v4h*)&As[r * LDH + kq] = cvt4_f16(f);
        }
        // stage B tile 64x32 from W (row n = output col)
        #pragma unroll
        for (int i = 0; i < 2; ++i) {
            int e = t + i * 256;          // 512 float4 slots
            int n = e >> 3, kq = (e & 7) * 4;
            f32x4 f = *(const f32x4*)&W[(size_t)(colBase + n) * CC + k0 + kq];
            *(v4h*)&Bs[n * LDH + kq] = cvt4_f16(f);
        }
        if (k0 + TK < CC)  // prefetch next A rows (global_prefetch_b8)
            __builtin_prefetch(&A[(size_t)(rowBase + (t >> 1)) * CC + k0 + TK], 0, 1);
        __syncthreads();

        v16h a = load_a_frag(As, wave * 16, lane);
        #pragma unroll
        for (int s = 0; s < 4; ++s) {
            v16h b = load_b_frag(Bs, s * 16, lane);
            acc[s] = __builtin_amdgcn_wmma_f32_16x16x32_f16(
                false, a, false, b, (short)0, acc[s], false, false);
        }
        __syncthreads();
    }

    // epilogue
    const int mrow0 = rowBase + wave * 16 + ((lane >> 4) << 3);
    #pragma unroll
    for (int s = 0; s < 4; ++s) {
        const int col = colBase + s * 16 + (lane & 15);
        const int c_g = col - cg * CC;
        const int h = c_g >> 6, hd = c_g & 63;
        const float bval = bias[col];
        float inv_scale = 0.f, p = 0.f;
        if (cg <= 1) {
            inv_scale = 1.0f / softplus_f(scale_p[c_g]);
            p = 1.0f + 4.0f * sigmoid_f(power_p[h * HD + hd]);
        }
        #pragma unroll
        for (int j = 0; j < 8; ++j) {
            const int m = mrow0 + j;
            const int b = m >> 9, n = m & (NN - 1);
            float val = acc[s][j] + bval;
            if (cg <= 1) {
                if (cg == 1) val += pe[(size_t)n * CC + c_g];
                val *= inv_scale;
                float pos = val > 0.f ? powf(val, p) : 0.f;
                float neg = val < 0.f ? powf(-val, p) : 0.f;
                size_t o = (((size_t)(b * HH + h) * NN + n) * 128) + hd;
                float* dst = (cg == 0) ? q_feat : k_feat;
                dst[o] = pos;
                dst[o + 64] = neg;
            } else if (cg == 2) {
                v_buf[(((size_t)(b * HH + h) * NN + n) * HD) + hd] = val;
            } else {
                g_buf[(size_t)m * CC + c_g] = val;
            }
        }
    }
}

// ---------------------------------------------------------------------------
// K2: per (b,h) compute k_mean[128] and kv[128][64] = sum_n kf[n][d]*v[n][e]/N
//     (LDS tiles filled with GLOBAL_LOAD_ASYNC_TO_LDS_B128)
// ---------------------------------------------------------------------------
__global__ __launch_bounds__(256)
void kv_reduce_kernel(const float* __restrict__ k_feat, const float* __restrict__ v_buf,
                      float* __restrict__ kmean, float* __restrict__ kv) {
    const int bh = blockIdx.x;
    const float* kf = k_feat + (size_t)bh * NN * 128;
    const float* vv = v_buf + (size_t)bh * NN * HD;
    __shared__ float kfs[64 * 128];
    __shared__ float vs[64 * 64];
    const int t = threadIdx.x;
    const int d = t >> 1;
    const int e0 = (t & 1) * 32;

    float accv[32] = {};
    float ksum = 0.f;

    for (int n0 = 0; n0 < NN; n0 += 64) {
        #pragma unroll
        for (int i = 0; i < 8; ++i) {    // 2048 float4 slots
            int q = (t + i * 256) * 4;
            async_copy_b128(&kf[(size_t)n0 * 128 + q], &kfs[q]);
        }
        #pragma unroll
        for (int i = 0; i < 4; ++i) {    // 1024 float4 slots
            int q = (t + i * 256) * 4;
            async_copy_b128(&vv[(size_t)n0 * 64 + q], &vs[q]);
        }
        async_copy_wait();
        __syncthreads();
        for (int nn = 0; nn < 64; ++nn) {
            const float kval = kfs[nn * 128 + d];
            ksum += kval;
            const float* vrow = &vs[nn * 64 + e0];
            #pragma unroll
            for (int i = 0; i < 32; ++i) accv[i] = fmaf(kval, vrow[i], accv[i]);
        }
        __syncthreads();
    }
    const float invN = 1.0f / (float)NN;
    float* kvo = kv + (size_t)bh * 128 * 64 + (size_t)d * 64 + e0;
    #pragma unroll
    for (int i = 0; i < 32; ++i) kvo[i] = accv[i] * invN;
    if ((t & 1) == 0) kmean[(size_t)bh * 128 + d] = ksum * invN;
}

// ---------------------------------------------------------------------------
// K3: per (b,h,n): z normalizers + x_sim/x_opp = q @ kv, write xa into attn
// ---------------------------------------------------------------------------
__global__ __launch_bounds__(256)
void attn_apply_kernel(const float* __restrict__ q_feat, const float* __restrict__ kmean,
                       const float* __restrict__ kv, float* __restrict__ attn) {
    const int bh = blockIdx.x;
    const int b = bh / HH, h = bh % HH;
    __shared__ float kvs[128 * 64];
    __shared__ float kms[128];
    const int t = threadIdx.x;
    #pragma unroll
    for (int i = 0; i < 8; ++i) {        // 2048 float4 slots
        int q = (t + i * 256) * 4;
        async_copy_b128(&kv[(size_t)bh * 8192 + q], &kvs[q]);
    }
    if (t < 32) async_copy_b128(&kmean[(size_t)bh * 128 + t * 4], &kms[t * 4]);
    async_copy_wait();
    __syncthreads();

    for (int rep = 0; rep < 2; ++rep) {
        const int n = t + rep * 256;
        const float* qf = q_feat + ((size_t)bh * NN + n) * 128;
        float qv[128];
        #pragma unroll 8
        for (int d = 0; d < 128; ++d) qv[d] = qf[d];
        float zs = 1e-6f, zo = 1e-6f;
        #pragma unroll 8
        for (int d = 0; d < 128; ++d) {
            zs = fmaf(qv[d], kms[d], zs);
            zo = fmaf(qv[(d + 64) & 127], kms[d], zo);
        }
        zs = 1.0f / zs;
        zo = 1.0f / zo;
        float* outp = attn + ((size_t)b * NN + n) * CC + h * HD;
        for (int e = 0; e < 32; ++e) {
            float xs = 0.f, xo = 0.f;
            #pragma unroll 8
            for (int d = 0; d < 128; ++d) {
                xs = fmaf(qv[d], kvs[d * 64 + e], xs);
                xo = fmaf(qv[(d + 64) & 127], kvs[d * 64 + 32 + e], xo);
            }
            outp[e] = xs * zs;
            outp[32 + e] = xo * zo;
        }
    }
}

// ---------------------------------------------------------------------------
// K4: depthwise 5x5x5 conv over the 8x8x8 token grid per (b,h).  Full v tile
// (512x64 f32) + weights live in LDS (~160 KB; WGP has 320 KB), filled with
// async B128 copies.  Adds the scrambled-reshape conv output into attn.
// ---------------------------------------------------------------------------
__global__ __launch_bounds__(256)
void dwconv_kernel(const float* __restrict__ v_buf, const float* __restrict__ w,
                   const float* __restrict__ wb, float* __restrict__ attn) {
    extern __shared__ float sm[];
    float* vsm = sm;            // 512*64
    float* wsm = sm + 512 * 64; // 64*125
    float* bsm = wsm + 64 * 125;

    const int bh = blockIdx.x;
    const int b = bh / HH, h = bh % HH;
    const int t = threadIdx.x;
    const float* vp = v_buf + (size_t)bh * NN * HD;
    #pragma unroll
    for (int i = 0; i < 32; ++i) {       // 8192 float4 slots
        int q = (t + i * 256) * 4;
        async_copy_b128(&vp[q], &vsm[q]);
    }
    for (int q4 = t; q4 < 2000; q4 += 256)   // 64*125 = 8000 floats
        async_copy_b128(&w[q4 * 4], &wsm[q4 * 4]);
    if (t < 16) async_copy_b128(&wb[t * 4], &bsm[t * 4]);
    async_copy_wait();
    __syncthreads();

    const int hd = t & 63;
    for (int sp = t >> 6; sp < NN; sp += 4) {
        const int dz = sp >> 6, dy = (sp >> 3) & 7, dx = sp & 7;
        float acc = bsm[hd];
        for (int kz = 0; kz < 5; ++kz) {
            const int z = dz + kz - 2;
            if (z < 0 || z > 7) continue;
            for (int ky = 0; ky < 5; ++ky) {
                const int yy = dy + ky - 2;
                if (yy < 0 || yy > 7) continue;
                #pragma unroll
                for (int kx = 0; kx < 5; ++kx) {
                    const int xx = dx + kx - 2;
                    if (xx < 0 || xx > 7) continue;
                    const int sp2 = (z * 8 + yy) * 8 + xx;
                    acc = fmaf(vsm[sp2 * 64 + hd], wsm[hd * 125 + (kz * 5 + ky) * 5 + kx], acc);
                }
            }
        }
        const size_t f = (size_t)h * (NN * HD) + (size_t)sp * HD + hd;
        attn[(size_t)b * (NN * CC) + f] += acc;   // (B,N,C) flat == b*196608 + f
    }
}

// ---------------------------------------------------------------------------
// K5: out = ((xa+vc)*g) @ Wproj^T + bproj   (WMMA f16, gate fused into A stage)
// ---------------------------------------------------------------------------
__global__ __launch_bounds__(256)
void proj_gemm_kernel(const float* __restrict__ attn, const float* __restrict__ g_buf,
                      const float* __restrict__ W, const float* __restrict__ bias,
                      float* __restrict__ out) {
    __shared__ _Float16 As[TM * LDH];
    __shared__ _Float16 Bs[TN * LDH];

    const int t = threadIdx.x;
    const int lane = t & 31;
    const int wave = t >> 5;
    const int rowBase = blockIdx.y * TM;
    const int colBase = blockIdx.x * TN;

    v8f acc[4] = {};

    for (int k0 = 0; k0 < CC; k0 += TK) {
        #pragma unroll
        for (int i = 0; i < 4; ++i) {
            int e = t + i * 256;
            int r = e >> 3, kq = (e & 7) * 4;
            size_t idx = (size_t)(rowBase + r) * CC + k0 + kq;
            f32x4 a4 = *(const f32x4*)&attn[idx];
            f32x4 g4 = *(const f32x4*)&g_buf[idx];
            *(v4h*)&As[r * LDH + kq] = cvt4_f16(a4 * g4);
        }
        #pragma unroll
        for (int i = 0; i < 2; ++i) {
            int e = t + i * 256;
            int n = e >> 3, kq = (e & 7) * 4;
            f32x4 f = *(const f32x4*)&W[(size_t)(colBase + n) * CC + k0 + kq];
            *(v4h*)&Bs[n * LDH + kq] = cvt4_f16(f);
        }
        __syncthreads();
        v16h a = load_a_frag(As, wave * 16, lane);
        #pragma unroll
        for (int s = 0; s < 4; ++s) {
            v16h b = load_b_frag(Bs, s * 16, lane);
            acc[s] = __builtin_amdgcn_wmma_f32_16x16x32_f16(
                false, a, false, b, (short)0, acc[s], false, false);
        }
        __syncthreads();
    }

    const int mrow0 = rowBase + wave * 16 + ((lane >> 4) << 3);
    #pragma unroll
    for (int s = 0; s < 4; ++s) {
        const int col = colBase + s * 16 + (lane & 15);
        const float bval = bias[col];
        #pragma unroll
        for (int j = 0; j < 8; ++j) {
            const int m = mrow0 + j;
            out[(size_t)m * CC + col] = acc[s][j] + bval;
        }
    }
}

// ---------------------------------------------------------------------------
extern "C" void kernel_launch(void* const* d_in, const int* in_sizes, int n_in,
                              void* d_out, int out_size, void* d_ws, size_t ws_size,
                              hipStream_t stream) {
    const float* x       = (const float*)d_in[0];
    const float* y       = (const float*)d_in[1];
    const float* Wqkvg   = (const float*)d_in[2];
    const float* bqkvg   = (const float*)d_in[3];
    const float* Wproj   = (const float*)d_in[4];
    const float* bproj   = (const float*)d_in[5];
    const float* dwc_w   = (const float*)d_in[6];
    const float* dwc_b   = (const float*)d_in[7];
    const float* power_p = (const float*)d_in[8];
    const float* scale_p = (const float*)d_in[9];
    const float* pe      = (const float*)d_in[10];
    float* out = (float*)d_out;

    float* ws = (float*)d_ws;
    const size_t featSz = (size_t)BB * HH * NN * 128;   // 12,582,912
    const size_t vSz    = (size_t)BB * HH * NN * HD;    //  6,291,456
    const size_t gSz    = (size_t)BB * NN * CC;         //  6,291,456
    float* q_feat = ws;
    float* k_feat = q_feat + featSz;
    float* v_buf  = k_feat + featSz;
    float* g_buf  = v_buf + vSz;
    float* kmean  = g_buf + gSz;                        // 24,576
    float* kv     = kmean + (size_t)BB * HH * 128;      // 1,572,864
    float* attn   = kv + (size_t)BB * HH * 128 * 64;    // 6,291,456

    dim3 g1(QKV_COLS / TN, M_TOT / TM);   // 24 x 128
    qkv_gemm_kernel<<<g1, 256, 0, stream>>>(x, y, Wqkvg, bqkvg, power_p, scale_p, pe,
                                            q_feat, k_feat, v_buf, g_buf);

    kv_reduce_kernel<<<BB * HH, 256, 0, stream>>>(k_feat, v_buf, kmean, kv);

    attn_apply_kernel<<<BB * HH, 256, 0, stream>>>(q_feat, kmean, kv, attn);

    const size_t convLds = (size_t)(512 * 64 + 64 * 125 + 64) * sizeof(float); // ~160 KB
    dwconv_kernel<<<BB * HH, 256, convLds, stream>>>(v_buf, dwc_w, dwc_b, attn);

    dim3 g5(CC / TN, M_TOT / TM);         // 6 x 128
    proj_gemm_kernel<<<g5, 256, 0, stream>>>(attn, g_buf, Wproj, bproj, out);
}
